// Attn_90726889161129
// MI455X (gfx1250) — matrix operand order repaired and checked by reading, not compiled, not run on hardware
//
#include <hip/hip_runtime.h>

#ifndef USE_TDM
#define USE_TDM 1
#endif

typedef __attribute__((ext_vector_type(16))) __bf16 v16bf;
typedef __attribute__((ext_vector_type(8)))  float  v8f;
typedef __attribute__((ext_vector_type(4)))  unsigned int u32x4;
typedef __attribute__((ext_vector_type(8)))  int i32x8;
typedef __attribute__((ext_vector_type(4)))  int i32x4;

union FragU { unsigned int u[8]; v16bf v; };

__device__ __forceinline__ unsigned short f32_to_bf16(float x) {
    unsigned int u = __float_as_uint(x);
    u += 0x7FFFu + ((u >> 16) & 1u);           // round-to-nearest-even
    return (unsigned short)(u >> 16);
}
__device__ __forceinline__ float bf16_to_f32(unsigned short h) {
    return __uint_as_float(((unsigned int)h) << 16);
}

// A-fragment (16x32 bf16, M x K) from LDS row-major [rows][pitch].
__device__ __forceinline__ v16bf frag_a(const unsigned short* lds, int row0, int pitch, int lane) {
    const unsigned short* p = lds + (row0 + (lane & 15)) * pitch;
    const int half8 = (lane >> 4) * 8;
    FragU f;
#pragma unroll
    for (int g = 0; g < 8; ++g) {
        int kb = ((g < 4) ? (g * 2) : (16 + (g - 4) * 2)) + half8;
        f.u[g] = *(const unsigned int*)(p + kb);
    }
    return f.v;
}
// B-fragment (32x16 bf16, K x N) from LDS stored transposed as [n][pitch(K)].
__device__ __forceinline__ v16bf frag_b(const unsigned short* lds, int col0, int pitch, int lane) {
    const unsigned short* p = lds + (col0 + (lane & 15)) * pitch;
    const int half8 = (lane >> 4) * 8;
    FragU f;
#pragma unroll
    for (int g = 0; g < 8; ++g)
        f.u[g] = *(const unsigned int*)(p + ((g < 4) ? (g * 2) : (16 + (g - 4) * 2)) + half8);
    return f.v;
}

#define WMMA_BF16(a, b, c) \
    __builtin_amdgcn_wmma_f32_16x16x32_bf16(false, (a), false, (b), (short)0, (c), false, false)

// ---------------------------------------------------------------------------
// Tensor Data Mover: load a 2D bf16 tile [rows x 32] (64B rows) from global to
// LDS with 16B padding after every row -> 80B LDS pitch (PITCH=40 ushorts).
// D# per CDNA5 ISA 8.3/8.4: data_size=1(2B), pad_enable, pad_interval=3
// (16 DWORDs), pad_amount=3 (4 DWORDs). tensor_dims set huge (no OOB clip).
// ---------------------------------------------------------------------------
__device__ __forceinline__ void tdm_load_tile32(unsigned lds_addr, const void* gp,
                                                unsigned ld_elems, unsigned rows)
{
#if USE_TDM
    unsigned long long ga = (unsigned long long)(uintptr_t)gp;
    u32x4 g0 = { 1u,                                   // count=1, user mode
                 lds_addr,                             // LDS byte address
                 (unsigned)(ga & 0xFFFFFFFFull),
                 (unsigned)((ga >> 32) & 0x01FFFFFFull) | (2u << 30) }; // type=2
    i32x8 g1 = { (int)((1u << 16) | (1u << 20) | (3u << 22) | (3u << 25)),
                 // ^ wg_mask=0, data_size=2B, pad_enable, interval=16dw, amount=4dw
                 0,                                    // atomic_barrier=0, dim0.lo16=0
                 0x10,                                 // dim0.hi16=0x10 (dim0=1M), dim1.lo16=0
                 0x10 | (32 << 16),                    // dim1.hi16 (dim1=1M), tile_dim0=32
                 (int)rows,                            // tile_dim1=rows, tile_dim2=0
                 (int)ld_elems,                        // tensor_dim0_stride (elements)
                 0, 0 };
    i32x4 z4 = { 0, 0, 0, 0 };
#if defined(__clang_major__) && (__clang_major__ >= 23)
    i32x8 z8 = { 0, 0, 0, 0, 0, 0, 0, 0 };
    __builtin_amdgcn_tensor_load_to_lds(g0, g1, z4, z4, z8, 0);
#else
    __builtin_amdgcn_tensor_load_to_lds(g0, g1, z4, z4, 0);
#endif
#endif
}
__device__ __forceinline__ unsigned lds_off_u32(const void* p) {
    return (unsigned)(uintptr_t)p;   // low 32 bits of generic shared ptr = LDS offset
}

// ---------------------------------------------------------------------------
// Generic tiled GEMM: Out(bf16)[M x N] = op(A[M x K] * B), optional tanh.
// AF32: A source f32 (convert while staging); AF32==0 -> bf16 A staged via TDM.
// BMODE: 0 -> B row-major [N x K]; 1 -> B row-major [K x N].
// Tiles 128x128x32, 8 waves (4x2), wave tile 32x64 (2x4 WMMA accumulators).
// ---------------------------------------------------------------------------
template<int AF32, int BMODE, int BF32, int DOTANH>
__global__ __launch_bounds__(256) void gemm_bf16_128x128(
    const void* __restrict__ Ap, long long sA, int lda,
    const void* __restrict__ Bp, long long sB, int ldb,
    unsigned short* __restrict__ Out, long long sO, int ldo, int K)
{
    constexpr int TM = 128, TN = 128, TK = 32, PITCH = 40;
    __shared__ __align__(16) unsigned short lA[TM * PITCH];
    __shared__ __align__(16) unsigned short lB[TN * PITCH];
    const int t = threadIdx.x, lane = t & 31, wave = t >> 5;
    const int wm = wave >> 1, wn = wave & 1;
    const int z = blockIdx.z;
    const long long m0 = (long long)blockIdx.x * TM;
    const int n0 = blockIdx.y * TN;

    const float*          Af = (const float*)Ap          + (long long)z * sA;
    const unsigned short* Ah = (const unsigned short*)Ap + (long long)z * sA;
    const float*          Bf = (const float*)Bp          + (long long)z * sB;
    const unsigned short* Bh = (const unsigned short*)Bp + (long long)z * sB;

    v8f acc[2][4] = {};
    for (int k0 = 0; k0 < K; k0 += TK) {
        // ---- stage A tile [TM x TK] ----
        if constexpr (AF32) {
            for (int idx = t; idx < TM * TK; idx += 256) {
                int r = idx >> 5, c = idx & 31;
                lA[r * PITCH + c] = f32_to_bf16(Af[(m0 + r) * (long long)lda + k0 + c]);
            }
        } else {
#if USE_TDM
            if (wave == 0)
                tdm_load_tile32(lds_off_u32(lA), Ah + m0 * (long long)lda + k0,
                                (unsigned)lda, (unsigned)TM);
#else
            for (int idx = t; idx < TM * TK; idx += 256) {
                int r = idx >> 5, c = idx & 31;
                lA[r * PITCH + c] = Ah[(m0 + r) * (long long)lda + k0 + c];
            }
#endif
        }
        // ---- stage B tile as BT [TN x TK] ----
        if constexpr (BMODE == 0) {
            for (int idx = t; idx < TN * TK; idx += 256) {
                int n = idx >> 5, c = idx & 31;
                long long src = (long long)(n0 + n) * ldb + k0 + c;
                if constexpr (BF32) lB[n * PITCH + c] = f32_to_bf16(Bf[src]);
                else                lB[n * PITCH + c] = Bh[src];
            }
        } else {
            for (int idx = t; idx < TN * TK; idx += 256) {
                int k = idx >> 7, n = idx & 127;
                long long src = (long long)(k0 + k) * ldb + n0 + n;
                if constexpr (BF32) lB[n * PITCH + k] = f32_to_bf16(Bf[src]);
                else                lB[n * PITCH + k] = Bh[src];
            }
        }
#if USE_TDM
        if constexpr (!AF32) {
            if (wave == 0) __builtin_amdgcn_s_wait_tensorcnt(0);
        }
#endif
        __syncthreads();
        v16bf a0 = frag_a(lA, wm * 32,      PITCH, lane);
        v16bf a1 = frag_a(lA, wm * 32 + 16, PITCH, lane);
#pragma unroll
        for (int j = 0; j < 4; ++j) {
            v16bf bj = frag_b(lB, wn * 64 + j * 16, PITCH, lane);
            acc[0][j] = WMMA_BF16(a0, bj, acc[0][j]);
            acc[1][j] = WMMA_BF16(a1, bj, acc[1][j]);
        }
        __syncthreads();
    }
    unsigned short* Op = Out + (long long)z * sO;
    const int half8 = (lane >> 4) * 8;
    const int nl = lane & 15;
#pragma unroll
    for (int i = 0; i < 2; ++i)
#pragma unroll
        for (int j = 0; j < 4; ++j)
#pragma unroll
            for (int e = 0; e < 8; ++e) {
                long long m = m0 + wm * 32 + i * 16 + half8 + e;
                int n = n0 + wn * 64 + j * 16 + nl;
                float v = acc[i][j][e];
                if constexpr (DOTANH) v = tanhf(v);
                Op[m * ldo + n] = f32_to_bf16(v);
            }
}

// ---------------------------------------------------------------------------
// Fused attention-score kernel: score[m] = sum_n w[n]*tanh( (A*B)[m,n] + bias[m,n] )
// A: bf16 [M x K] (TRANSA=1: source stored [K x M], staged transposed manually;
//                  TRANSA=0: staged via TDM)
// B: bf16 [K x N] row-major (N=256, full width per workgroup)
// Tiles 64x256x32, 8 waves (2x4), wave tile 32x64 (2x4 accumulators).
// ---------------------------------------------------------------------------
template<int TRANSA>
__global__ __launch_bounds__(256) void attn_score_64x256(
    const unsigned short* __restrict__ Ap, long long sA, int lda,
    const unsigned short* __restrict__ Bp, long long sB, int ldb,
    const unsigned short* __restrict__ biasP, long long sBias, int ldbias,
    const float* __restrict__ w, float* __restrict__ scores,
    int Lrows, int K)
{
    constexpr int TM = 64, TN = 256, TK = 32, PITCH = 40;
    __shared__ __align__(16) unsigned short lA[TM * PITCH];
    __shared__ __align__(16) unsigned short lB[TN * PITCH];
    __shared__ float sScore[TM];
    __shared__ float sW[TN];
    const int t = threadIdx.x, lane = t & 31, wave = t >> 5;
    const int wm = wave >> 2, wn = wave & 3;
    const int z = blockIdx.z;
    const int m0 = blockIdx.x * TM;
    const unsigned short* A    = Ap    + (long long)z * sA;
    const unsigned short* B    = Bp    + (long long)z * sB;
    const unsigned short* bias = biasP + (long long)z * sBias;
    if (t < TN) sW[t] = w[t];
    if (t < TM) sScore[t] = 0.f;

    v8f acc[2][4] = {};
    for (int k0 = 0; k0 < K; k0 += TK) {
        if constexpr (TRANSA) {
            for (int idx = t; idx < TK * TM; idx += 256) {
                int k = idx >> 6, m = idx & 63;
                lA[m * PITCH + k] = A[(long long)(k0 + k) * lda + m0 + m];
            }
        } else {
#if USE_TDM
            if (wave == 0)
                tdm_load_tile32(lds_off_u32(lA), A + (long long)m0 * lda + k0,
                                (unsigned)lda, (unsigned)TM);
#else
            for (int idx = t; idx < TM * TK; idx += 256) {
                int r = idx >> 5, c = idx & 31;
                lA[r * PITCH + c] = A[(long long)(m0 + r) * lda + k0 + c];
            }
#endif
        }
        for (int idx = t; idx < TK * TN; idx += 256) {
            int k = idx >> 8, n = idx & 255;
            lB[n * PITCH + k] = B[(long long)(k0 + k) * ldb + n];
        }
#if USE_TDM
        if constexpr (!TRANSA) {
            if (wave == 0) __builtin_amdgcn_s_wait_tensorcnt(0);
        }
#endif
        __syncthreads();
        v16bf a0 = frag_a(lA, wm * 32,      PITCH, lane);
        v16bf a1 = frag_a(lA, wm * 32 + 16, PITCH, lane);
#pragma unroll
        for (int j = 0; j < 4; ++j) {
            v16bf bj = frag_b(lB, wn * 64 + j * 16, PITCH, lane);
            acc[0][j] = WMMA_BF16(a0, bj, acc[0][j]);
            acc[1][j] = WMMA_BF16(a1, bj, acc[1][j]);
        }
        __syncthreads();
    }
    const int half8 = (lane >> 4) * 8;
    const int nl = lane & 15;
#pragma unroll
    for (int i = 0; i < 2; ++i)
#pragma unroll
        for (int j = 0; j < 4; ++j)
#pragma unroll
            for (int e = 0; e < 8; ++e) {
                int ml = wm * 32 + i * 16 + half8 + e;
                int n  = wn * 64 + j * 16 + nl;
                float bv = bf16_to_f32(bias[(long long)(m0 + ml) * ldbias + n]);
                float h  = tanhf(acc[i][j][e] + bv);
                atomicAdd(&sScore[ml], h * sW[n]);   // ds_add_f32
            }
    __syncthreads();
    if (t < TM) scores[(long long)z * Lrows + m0 + t] = sScore[t];
}

// ---------------------------------------------------------------------------
// Masked softmax (faithful: softmax(v*m), re-mask, renorm + eps) then
// out[b,d] = sum_l attn[l] * seq[b,l,d].  Requires L == D == blockDim == 512.
// ---------------------------------------------------------------------------
__global__ __launch_bounds__(512) void softmax_pool(
    const float* __restrict__ scores, const int* __restrict__ mask,
    const float* __restrict__ seq, float* __restrict__ out, int L, int D)
{
    __shared__ float red[512];
    __shared__ float attn[512];
    const int t = threadIdx.x;
    const int b = blockIdx.x;
    float mf = (float)mask[b * L + t];
    float x  = scores[b * L + t] * mf;
    red[t] = x; __syncthreads();
    for (int s = 256; s > 0; s >>= 1) { if (t < s) red[t] = fmaxf(red[t], red[t + s]); __syncthreads(); }
    float mx = red[0]; __syncthreads();
    float e = __expf(x - mx);
    red[t] = e; __syncthreads();
    for (int s = 256; s > 0; s >>= 1) { if (t < s) red[t] += red[t + s]; __syncthreads(); }
    float s1 = red[0]; __syncthreads();
    float r = (e / s1) * mf;
    red[t] = r; __syncthreads();
    for (int s = 256; s > 0; s >>= 1) { if (t < s) red[t] += red[t + s]; __syncthreads(); }
    float s2 = red[0];
    attn[t] = r / (s2 + 1e-13f);
    __syncthreads();
    float accv = 0.f;
    const float* sb = seq + (long long)b * L * D;
    for (int l = 0; l < L; ++l) accv += attn[l] * sb[(long long)l * D + t];
    out[b * D + t] = accv;
}

// ---------------------------------------------------------------------------
extern "C" void kernel_launch(void* const* d_in, const int* in_sizes, int n_in,
                              void* d_out, int out_size, void* d_ws, size_t ws_size,
                              hipStream_t stream)
{
    const float* S1    = (const float*)d_in[0];   // [B,L,D]
    const float* S2    = (const float*)d_in[1];   // [B,L,D]
    const int*   mask1 = (const int*)d_in[2];
    const int*   mask2 = (const int*)d_in[3];
    const float* W     = (const float*)d_in[4];   // [D,D]
    const float* Wv    = (const float*)d_in[5];   // [D,A]
    const float* Wq    = (const float*)d_in[6];   // [D,A]
    const float* whv   = (const float*)d_in[7];   // [A,1]
    const float* whq   = (const float*)d_in[8];   // [A,1]

    const int B = 64, L = 512, D = 512, A = 256;

    char* ws = (char*)d_ws;
    size_t off = 0;
    unsigned short* T1 = (unsigned short*)(ws + off); off += (size_t)B * L * D * 2; // 33.5 MB
    unsigned short* Vb = (unsigned short*)(ws + off); off += (size_t)B * L * A * 2; // 16.8 MB
    unsigned short* Qb = (unsigned short*)(ws + off); off += (size_t)B * L * A * 2; // 16.8 MB
    unsigned short* Cb = (unsigned short*)(ws + off); off += (size_t)B * L * L * 2; // 33.5 MB
    float* sv = (float*)(ws + off); off += (size_t)B * L * 4;
    float* sq = (float*)(ws + off);

    dim3 blk(256);

    // T1 = S1 @ W          (A f32, B [K x N] f32)
    gemm_bf16_128x128<1, 1, 1, 0><<<dim3(B * L / 128, D / 128, 1), blk, 0, stream>>>(
        S1, 0, D, W, 0, D, T1, 0, D, D);
    // V = S1 @ Wv
    gemm_bf16_128x128<1, 1, 1, 0><<<dim3(B * L / 128, A / 128, 1), blk, 0, stream>>>(
        S1, 0, D, Wv, 0, A, Vb, 0, A, D);
    // Q = S2 @ Wq
    gemm_bf16_128x128<1, 1, 1, 0><<<dim3(B * L / 128, A / 128, 1), blk, 0, stream>>>(
        S2, 0, D, Wq, 0, A, Qb, 0, A, D);
    // C = tanh(T1 @ S2^T)  per batch (A bf16 via TDM, B [N x K] f32 = S2 rows)
    gemm_bf16_128x128<0, 0, 1, 1><<<dim3(L / 128, L / 128, B), blk, 0, stream>>>(
        T1, (long long)L * D, D, S2, (long long)L * D, D, Cb, (long long)L * L, L, D);
    // scores_v[m] = sum_a w_hv[a] * tanh(V + C @ Q)[m,a]   (A = C via TDM)
    attn_score_64x256<0><<<dim3(L / 64, 1, B), blk, 0, stream>>>(
        Cb, (long long)L * L, L, Qb, (long long)L * A, A,
        Vb, (long long)L * A, A, whv, sv, L, L);
    // scores_q[m] = sum_a w_hq[a] * tanh(Q + C^T @ V)[m,a]
    attn_score_64x256<1><<<dim3(L / 64, 1, B), blk, 0, stream>>>(
        Cb, (long long)L * L, L, Vb, (long long)L * A, A,
        Qb, (long long)L * A, A, whq, sq, L, L);

    float* out = (float*)d_out;
    softmax_pool<<<dim3(B), dim3(512), 0, stream>>>(sv, mask1, S1, out, L, D);
    softmax_pool<<<dim3(B), dim3(512), 0, stream>>>(sq, mask2, S2, out + (size_t)B * D, L, D);
}
